// Attention_20074677141766
// MI455X (gfx1250) — matrix operand order repaired and checked by reading, not compiled, not run on hardware
//
#include <hip/hip_runtime.h>
#include <hip/hip_bf16.h>
#include <math.h>

typedef __attribute__((ext_vector_type(16))) _Float16 v16h;
typedef __attribute__((ext_vector_type(8)))  _Float16 v8h;
typedef __attribute__((ext_vector_type(8)))  float    v8f;
typedef __attribute__((ext_vector_type(4)))  unsigned int v4u;
typedef __attribute__((ext_vector_type(8)))  int v8i;
typedef __attribute__((ext_vector_type(4)))  int v4i;

#define WMMA_F16(a, b, c) \
  __builtin_amdgcn_wmma_f32_16x16x32_f16(false, (a), false, (b), (short)0, (c), false, false)

static constexpr int S_      = 49;
static constexpr int NWIN    = 2048;            // B*N = 8*256 windows
static constexpr int MROWS   = NWIN * S_;       // 100352 token rows

// ---------------- fp32 -> fp16 conversion (weights) ----------------
__global__ void f32_to_f16_kernel(const float* __restrict__ in,
                                  _Float16* __restrict__ out, int n) {
  int i = blockIdx.x * blockDim.x + threadIdx.x;
  if (i < n) out[i] = (_Float16)in[i];
}

// ---------------- LayerNorm over C=256, fp16 output ----------------
__global__ __launch_bounds__(256) void ln_f16_kernel(
    const float* __restrict__ X, const float* __restrict__ g,
    const float* __restrict__ b, _Float16* __restrict__ O) {
  const int row  = blockIdx.x * 8 + (threadIdx.x >> 5);
  const int lane = threadIdx.x & 31;
  const float* p = X + (size_t)row * 256 + lane * 8;
  float v[8];
  const float4 a0 = ((const float4*)p)[0];
  const float4 a1 = ((const float4*)p)[1];
  v[0]=a0.x; v[1]=a0.y; v[2]=a0.z; v[3]=a0.w;
  v[4]=a1.x; v[5]=a1.y; v[6]=a1.z; v[7]=a1.w;
  float s = 0.f;
  #pragma unroll
  for (int i = 0; i < 8; i++) s += v[i];
  #pragma unroll
  for (int o = 16; o >= 1; o >>= 1) s += __shfl_xor(s, o, 32);
  const float mu = s * (1.f / 256.f);
  float vs = 0.f;
  #pragma unroll
  for (int i = 0; i < 8; i++) { float d = v[i] - mu; vs += d * d; }
  #pragma unroll
  for (int o = 16; o >= 1; o >>= 1) vs += __shfl_xor(vs, o, 32);
  const float rstd = rsqrtf(vs * (1.f / 256.f) + 1e-5f);
  _Float16* op = O + (size_t)row * 256 + lane * 8;
  #pragma unroll
  for (int i = 0; i < 8; i++) {
    int c = lane * 8 + i;
    op[i] = (_Float16)((v[i] - mu) * rstd * g[c] + b[c]);
  }
}

// ---------------- Tensor Data Mover: stage 2-D fp16 tile -> LDS ----------------
// D# per CDNA5 ISA: group0 = {count, lds_addr, global_addr, type=2},
// group1 = {data_size=2B, tensor dims, tile dims, row stride}. 2-D tile: groups 2/3 zero.
__device__ inline void tdm_load_tile_2d(unsigned lds_off, const void* gptr,
                                        int rowlen, int nrows, int rowstride) {
  const unsigned long long ga = (unsigned long long)(uintptr_t)gptr;
  v4u g0;
  g0[0] = 1u;                                           // count=1, user descriptor
  g0[1] = lds_off;                                      // LDS byte address
  g0[2] = (unsigned)(ga & 0xffffffffu);                 // global_addr[31:0]
  g0[3] = (unsigned)((ga >> 32) & 0x1ffffffu) | (2u << 30);  // global_addr[56:32] | type=2
  v8i g1;
  g1[0] = (1 << 16);                                    // data_size code 1 = 2 bytes
  g1[1] = (int)((rowlen & 0xffff) << 16);               // tensor_dim0[15:0]
  g1[2] = (int)(((unsigned)rowlen >> 16) | ((unsigned)(nrows & 0xffff) << 16)); // dim0[31:16], dim1[15:0]
  g1[3] = (int)((rowlen & 0xffff) << 16);               // dim1[31:16]=0, tile_dim0=rowlen
  g1[4] = (nrows & 0xffff);                             // tile_dim1=nrows, tile_dim2=0
  g1[5] = rowstride;                                    // tensor_dim0_stride[31:0]
  g1[6] = 0;                                            // stride[47:32]=0, dim1_stride lo=0
  g1[7] = 0;
  v4i z4; z4[0]=0; z4[1]=0; z4[2]=0; z4[3]=0;
#if defined(__clang_major__) && (__clang_major__ >= 23)
  v8i z8; for (int i = 0; i < 8; i++) z8[i] = 0;
  __builtin_amdgcn_tensor_load_to_lds(g0, g1, z4, z4, z8, 0);
#else
  __builtin_amdgcn_tensor_load_to_lds(g0, g1, z4, z4, 0);
#endif
}

// ---------------- fragment helper (contiguous 16-bit A layout, k = 16*hf + s) ----------------
__device__ inline v16h load_a_frag(const _Float16* rowptr, int k0, int hf) {
  const v8h c1 = *(const v8h*)(rowptr + k0 + 16 * hf);
  const v8h c2 = *(const v8h*)(rowptr + k0 + 16 * hf + 8);
  v16h a;
  #pragma unroll
  for (int i = 0; i < 8; i++) { a[i] = c1[i]; a[i + 8] = c2[i]; }
  return a;
}

// ---------------- GEMM: C16[M,N] = f16(A16[M,K] @ W16[N,K]^T + bias) ----------------
// TDM stages the 32xK A-tile into LDS once per block; 8 waves -> 32x64 tile.
template <int K>
__global__ __launch_bounds__(256) void gemm_f16_kernel(
    const _Float16* __restrict__ A, const _Float16* __restrict__ W,
    const float* __restrict__ bias, _Float16* __restrict__ Cout,
    int N, int act) {
  __shared__ _Float16 ldsA[32 * K];
  const int wid = threadIdx.x >> 5, lane = threadIdx.x & 31;
  const int hf = lane >> 4, r = lane & 15;
  const int m0 = blockIdx.y * 32 + (wid >> 2) * 16;
  const int n0 = blockIdx.x * 64 + (wid & 3) * 16;
  if (threadIdx.x < 32) {
    tdm_load_tile_2d((unsigned)(uintptr_t)(void*)ldsA,
                     A + (size_t)blockIdx.y * 32 * K, K, 32, K);
    __builtin_amdgcn_s_wait_tensorcnt(0);
  }
  __syncthreads();
  v8f acc;
  #pragma unroll
  for (int j = 0; j < 8; j++) acc[j] = 0.f;
  const _Float16* arow = ldsA + ((wid >> 2) * 16 + r) * K;   // LDS (ds_load_b128)
  const _Float16* wrow = W + (size_t)(n0 + r) * K;           // global (L2-resident)
  #pragma unroll
  for (int k0 = 0; k0 < K; k0 += 32) {
    const v16h af = load_a_frag(arow, k0, hf);
    const v16h bf = *(const v16h*)(wrow + k0 + 16 * hf);
    acc = WMMA_F16(af, bf, acc);
  }
  const int n = n0 + r;
  const float bn = bias[n];
  #pragma unroll
  for (int j = 0; j < 8; j++) {
    const int m = m0 + j + 8 * hf;
    float v = acc[j] + bn;
    if (act) v = 0.5f * v * (1.f + erff(v * 0.70710678118654752f));
    Cout[(size_t)m * N + n] = (_Float16)v;
  }
}

// ---------------- GEMM + bias + residual, fp32 output ----------------
template <int K>
__global__ __launch_bounds__(256) void gemm_res_f32_kernel(
    const _Float16* __restrict__ A, const _Float16* __restrict__ W,
    const float* __restrict__ bias, const float* __restrict__ R,
    float* __restrict__ Cout, int N) {
  __shared__ _Float16 ldsA[32 * K];
  const int wid = threadIdx.x >> 5, lane = threadIdx.x & 31;
  const int hf = lane >> 4, r = lane & 15;
  const int m0 = blockIdx.y * 32 + (wid >> 2) * 16;
  const int n0 = blockIdx.x * 64 + (wid & 3) * 16;
  if (threadIdx.x < 32) {
    tdm_load_tile_2d((unsigned)(uintptr_t)(void*)ldsA,
                     A + (size_t)blockIdx.y * 32 * K, K, 32, K);
    __builtin_amdgcn_s_wait_tensorcnt(0);
  }
  __syncthreads();
  v8f acc;
  #pragma unroll
  for (int j = 0; j < 8; j++) acc[j] = 0.f;
  const _Float16* arow = ldsA + ((wid >> 2) * 16 + r) * K;
  const _Float16* wrow = W + (size_t)(n0 + r) * K;
  #pragma unroll
  for (int k0 = 0; k0 < K; k0 += 32) {
    const v16h af = load_a_frag(arow, k0, hf);
    const v16h bf = *(const v16h*)(wrow + k0 + 16 * hf);
    acc = WMMA_F16(af, bf, acc);
  }
  const int n = n0 + r;
  const float bn = bias[n];
  #pragma unroll
  for (int j = 0; j < 8; j++) {
    const int m = m0 + j + 8 * hf;
    Cout[(size_t)m * N + n] = acc[j] + bn + R[(size_t)m * N + n];
  }
}

// ---------------- windowed attention: 1 block per window, 1 wave per head ----------------
__global__ __launch_bounds__(256) void attn_kernel(
    const _Float16* __restrict__ qkv, const float* __restrict__ bias_table,
    _Float16* __restrict__ out) {
  __shared__ _Float16 lds_attn[8][64 * 64];   // 64 KB: f16 softmax probs per head
  const int win = blockIdx.x;
  const int n = win & 255;
  const int wr = n >> 4, wc = n & 15;
  const int head = threadIdx.x >> 5;
  const int lane = threadIdx.x & 31;
  const int hf = lane >> 4, r = lane & 15;
  const size_t rowbase = (size_t)win * 49;
  const _Float16* qb = qkv + rowbase * 768 + head * 32;
  const _Float16* kb = qb + 256;
  const _Float16* vb = qb + 512;

  // ---- Q @ K^T : 4x4 tiles, hd=32 = one WMMA K-step ----
  v16h kf[4];
  #pragma unroll
  for (int nt = 0; nt < 4; nt++) {
    const int t = 16 * nt + r;
    if (t < 49) kf[nt] = *(const v16h*)(kb + (size_t)t * 768 + 16 * hf);
    else {
      v16h z;
      #pragma unroll
      for (int e = 0; e < 16; e++) z[e] = (_Float16)0.f;
      kf[nt] = z;
    }
  }
  v8f acc[4][4];
  #pragma unroll
  for (int mt = 0; mt < 4; mt++) {
    const int s = 16 * mt + r;
    v16h qf;
    if (s < 49) qf = load_a_frag(qb + (size_t)s * 768, 0, hf);
    else {
      #pragma unroll
      for (int e = 0; e < 16; e++) qf[e] = (_Float16)0.f;
    }
    #pragma unroll
    for (int nt = 0; nt < 4; nt++) {
      v8f z;
      #pragma unroll
      for (int j = 0; j < 8; j++) z[j] = 0.f;
      acc[mt][nt] = WMMA_F16(qf, kf[nt], z);
    }
  }

  // ---- scale + rel-pos bias + shift mask + softmax ----
  const float scale = 0.17677669529663689f;   // 32^-0.5
  #pragma unroll
  for (int mt = 0; mt < 4; mt++) {
    #pragma unroll
    for (int j = 0; j < 8; j++) {
      const int s = 16 * mt + j + 8 * hf;     // uniform within 16-lane shuffle group
      const int sh = s / 7, sw = s - sh * 7;
      float vals[4];
      float rmax = -INFINITY;
      #pragma unroll
      for (int nt = 0; nt < 4; nt++) {
        const int t = 16 * nt + r;
        float val = -INFINITY;
        if (s < 49 && t < 49) {
          const int th = t / 7, tw = t - th * 7;
          const bool msk = (wr == 15 && ((sh < 4) != (th < 4))) ||
                           (wc == 15 && ((sw < 4) != (tw < 4)));
          if (!msk) {
            const int rel = (sh - th + 6) * 13 + (sw - tw + 6);
            val = acc[mt][nt][j] * scale + bias_table[rel * 8 + head];
          }
        }
        vals[nt] = val;
        rmax = fmaxf(rmax, val);
      }
      #pragma unroll
      for (int o = 8; o >= 1; o >>= 1) rmax = fmaxf(rmax, __shfl_xor(rmax, o, 32));
      float e[4], rs = 0.f;
      #pragma unroll
      for (int nt = 0; nt < 4; nt++) { e[nt] = __expf(vals[nt] - rmax); rs += e[nt]; }
      #pragma unroll
      for (int o = 8; o >= 1; o >>= 1) rs += __shfl_xor(rs, o, 32);
      const float inv = 1.f / rs;
      #pragma unroll
      for (int nt = 0; nt < 4; nt++) {
        _Float16 o16 = (_Float16)0.f;
        if (s < 49) o16 = (_Float16)(e[nt] * inv);
        lds_attn[head][s * 64 + 16 * nt + r] = o16;
      }
    }
  }

  // ---- P @ V : K = 64 (2 steps), N = 32 (2 tiles) ----
  #pragma unroll
  for (int dt = 0; dt < 2; dt++) {
    const int d = 16 * dt + r;
    v16h vf[2];
    #pragma unroll
    for (int ks = 0; ks < 2; ks++) {
      v16h f;
      #pragma unroll
      for (int e = 0; e < 16; e++) {
        const int t = 32 * ks + 16 * hf + e;
        f[e] = (t < 49) ? vb[(size_t)t * 768 + d] : (_Float16)0.f;
      }
      vf[ks] = f;
    }
    #pragma unroll
    for (int mt = 0; mt < 4; mt++) {
      v8f o;
      #pragma unroll
      for (int j = 0; j < 8; j++) o[j] = 0.f;
      #pragma unroll
      for (int ks = 0; ks < 2; ks++) {
        const int m = 16 * mt + r;
        const _Float16* lp = &lds_attn[head][m * 64 + 32 * ks];
        const v8h c1 = *(const v8h*)(lp + 16 * hf);
        const v8h c2 = *(const v8h*)(lp + 16 * hf + 8);
        v16h af;
        #pragma unroll
        for (int i = 0; i < 8; i++) { af[i] = c1[i]; af[i + 8] = c2[i]; }
        o = WMMA_F16(af, vf[ks], o);
      }
      #pragma unroll
      for (int j = 0; j < 8; j++) {
        const int s = 16 * mt + j + 8 * hf;
        if (s < 49)
          out[(rowbase + s) * 256 + head * 32 + d] = (_Float16)o[j];
      }
    }
  }
}

// ---------------- host-side orchestration ----------------
extern "C" void kernel_launch(void* const* d_in, const int* in_sizes, int n_in,
                              void* d_out, int out_size, void* d_ws, size_t ws_size,
                              hipStream_t stream) {
  (void)in_sizes; (void)n_in; (void)out_size; (void)ws_size;
  const float* x       = (const float*)d_in[0];
  const float* n1g     = (const float*)d_in[2];
  const float* n1b     = (const float*)d_in[3];
  const float* qkv_w   = (const float*)d_in[4];
  const float* qkv_b   = (const float*)d_in[5];
  const float* proj_w  = (const float*)d_in[6];
  const float* proj_b  = (const float*)d_in[7];
  const float* n2g     = (const float*)d_in[8];
  const float* n2b     = (const float*)d_in[9];
  const float* w1      = (const float*)d_in[10];
  const float* b1      = (const float*)d_in[11];
  const float* w2      = (const float*)d_in[12];
  const float* b2      = (const float*)d_in[13];
  const float* btab    = (const float*)d_in[14];
  float* out = (float*)d_out;

  char* ws = (char*)d_ws;
  const size_t SZ_X2   = (size_t)MROWS * 256 * 4;
  const size_t SZ_H16  = (size_t)MROWS * 256 * 2;
  const size_t SZ_BIG  = (size_t)MROWS * 1024 * 2;
  float*    x2    = (float*)   (ws);
  _Float16* h16   = (_Float16*)(ws + SZ_X2);
  _Float16* big   = (_Float16*)(ws + SZ_X2 + SZ_H16);
  _Float16* attn16= (_Float16*)(ws + SZ_X2 + SZ_H16 + SZ_BIG);
  char* wbase = ws + SZ_X2 + SZ_H16 + SZ_BIG + SZ_H16;
  _Float16* wq16 = (_Float16*)(wbase);
  _Float16* wp16 = (_Float16*)(wbase + 768 * 256 * 2);
  _Float16* w116 = (_Float16*)(wbase + 768 * 256 * 2 + 256 * 256 * 2);
  _Float16* w216 = (_Float16*)(wbase + 768 * 256 * 2 + 256 * 256 * 2 + 1024 * 256 * 2);

  f32_to_f16_kernel<<<(768 * 256 + 255) / 256, 256, 0, stream>>>(qkv_w, wq16, 768 * 256);
  f32_to_f16_kernel<<<(256 * 256 + 255) / 256, 256, 0, stream>>>(proj_w, wp16, 256 * 256);
  f32_to_f16_kernel<<<(1024 * 256 + 255) / 256, 256, 0, stream>>>(w1, w116, 1024 * 256);
  f32_to_f16_kernel<<<(256 * 1024 + 255) / 256, 256, 0, stream>>>(w2, w216, 256 * 1024);

  // LN1 -> h16
  ln_f16_kernel<<<MROWS / 8, 256, 0, stream>>>(x, n1g, n1b, h16);
  // QKV GEMM (K=256): TDM-staged A
  gemm_f16_kernel<256><<<dim3(768 / 64, MROWS / 32), 256, 0, stream>>>(
      h16, wq16, qkv_b, big, 768, 0);
  // windowed attention
  attn_kernel<<<NWIN, 256, 0, stream>>>(big, btab, attn16);
  // proj + residual(x) -> x2
  gemm_res_f32_kernel<256><<<dim3(256 / 64, MROWS / 32), 256, 0, stream>>>(
      attn16, wp16, proj_b, x, x2, 256);
  // LN2 -> h16
  ln_f16_kernel<<<MROWS / 8, 256, 0, stream>>>(x2, n2g, n2b, h16);
  // MLP1 + GELU (K=256)
  gemm_f16_kernel<256><<<dim3(1024 / 64, MROWS / 32), 256, 0, stream>>>(
      h16, w116, b1, big, 1024, 1);
  // MLP2 + residual(x2) -> out (K=1024)
  gemm_res_f32_kernel<1024><<<dim3(256 / 64, MROWS / 32), 256, 0, stream>>>(
      big, w216, b2, x2, out, 256);
}
